// Expert_Choice_9732395892786
// MI455X (gfx1250) — compile-verified, matrix operand order checked by reading, not run on hardware
//
#include <hip/hip_runtime.h>
#include <hip/hip_bf16.h>

// ---------------------------------------------------------------------------
// Expert-choice MoE for MI455X (gfx1250, wave32, WMMA + async-to-LDS staging)
// ---------------------------------------------------------------------------

typedef int      v4i  __attribute__((ext_vector_type(4)));
typedef _Float16 v4h  __attribute__((ext_vector_type(4)));
typedef _Float16 v8h  __attribute__((ext_vector_type(8)));
typedef _Float16 v16h __attribute__((ext_vector_type(16)));
typedef float    v8f  __attribute__((ext_vector_type(8)));

constexpr int F  = 128;   // features
constexpr int H  = 256;   // hidden
constexpr int O  = 64;    // output
constexpr int E  = 16;    // experts
constexpr int TM = 64;    // token tile per block
constexpr int KC = 64;    // hidden chunk

constexpr int XS_STRIDE = F  + 8;   // 136 halfwords -> 272B rows (16B aligned, breaks 64-bank stride)
constexpr int W1_STRIDE = F  + 8;
constexpr int HC_STRIDE = KC + 8;   // 72 halfwords -> 144B rows
constexpr int W2_STRIDE = KC + 8;

#if defined(__gfx1250__) && __has_builtin(__builtin_amdgcn_global_load_async_to_lds_b128) && \
    __has_builtin(__builtin_amdgcn_s_wait_asynccnt)
#define USE_ASYNC_LDS 1
#else
#define USE_ASYNC_LDS 0
#endif

#if USE_ASYNC_LDS
typedef __attribute__((address_space(1))) v4i GB128;   // global 16B chunk
typedef __attribute__((address_space(3))) v4i LB128;   // LDS 16B chunk
// 16B global -> LDS async copy (ASYNCcnt-tracked). LDS address is the low 32 bits of the pointer.
__device__ __forceinline__ void asyncCopy16(const void* g, void* l) {
  GB128* gp = (GB128*)(uintptr_t)g;
  LB128* lp = (LB128*)(unsigned)(uintptr_t)l;
  __builtin_amdgcn_global_load_async_to_lds_b128(gp, lp, 0, 0);
}
#endif

__device__ __forceinline__ unsigned sortKey(float s) {
  unsigned u = __float_as_uint(s);
  return (u & 0x80000000u) ? ~u : (u | 0x80000000u);   // monotone: bigger float -> bigger key
}

__device__ __forceinline__ void atomAddF(float* p, float v) {
  __hip_atomic_fetch_add(p, v, __ATOMIC_RELAXED, __HIP_MEMORY_SCOPE_AGENT);
}

// Load a 16x32 f16 A/B fragment from row-major LDS per CDNA5 WMMA striping:
// lane (l&15) = M (or N), lane half hh picks the K sub-bands {hh*8..} and {16+hh*8..}.
__device__ __forceinline__ v16h ldfrag(const _Float16* base, int row, int stride,
                                       int k0, int hh) {
  const _Float16* p = base + row * stride + k0 + hh * 8;
  v8h lo = *(const v8h*)(p);
  v8h hi = *(const v8h*)(p + 16);
  return __builtin_shufflevector(lo, hi, 0,1,2,3,4,5,6,7,8,9,10,11,12,13,14,15);
}

// ------------------- one-time weight conversion f32 -> f16 -----------------
__global__ __launch_bounds__(256)
void moe_cvt_weights(const float* __restrict__ W1, const float* __restrict__ W2,
                     _Float16* __restrict__ W1h, _Float16* __restrict__ W2h,
                     int n1q, int n2q) {   // counts in float4 units
  int i = blockIdx.x * blockDim.x + threadIdx.x;
  if (i < n1q) {
    float4 x = ((const float4*)W1)[i];
    v4h h = { (_Float16)x.x, (_Float16)x.y, (_Float16)x.z, (_Float16)x.w };
    ((v4h*)W1h)[i] = h;
  } else if (i < n1q + n2q) {
    int j = i - n1q;
    float4 x = ((const float4*)W2)[j];
    v4h h = { (_Float16)x.x, (_Float16)x.y, (_Float16)x.z, (_Float16)x.w };
    ((v4h*)W2h)[j] = h;
  }
}

// ------------------------------ gate ---------------------------------------
__global__ __launch_bounds__(256)
void moe_gate(const float* __restrict__ feat, const float* __restrict__ Wg,
              const float* __restrict__ bg, float* __restrict__ scoresT, int B) {
  __shared__ float WgS[E][F];
  __shared__ float bgS[E];
  for (int i = threadIdx.x; i < E * F; i += blockDim.x) WgS[i / F][i % F] = Wg[i];
  if (threadIdx.x < E) bgS[threadIdx.x] = bg[threadIdx.x];
  __syncthreads();

  int t = blockIdx.x * blockDim.x + threadIdx.x;
  if (t >= B) return;

  float acc[E];
#pragma unroll
  for (int e = 0; e < E; ++e) acc[e] = bgS[e];

  const float* fr = feat + (size_t)t * F;
#pragma unroll 4
  for (int f = 0; f < F; f += 4) {
    float4 x = *(const float4*)(fr + f);
#pragma unroll
    for (int e = 0; e < E; ++e) {
      acc[e] += x.x * WgS[e][f]     + x.y * WgS[e][f + 1]
              + x.z * WgS[e][f + 2] + x.w * WgS[e][f + 3];
    }
  }
#pragma unroll
  for (int e = 0; e < E; ++e) scoresT[(size_t)e * B + t] = acc[e];
}

// -------------------- per-expert exact top-M threshold ---------------------
// 4-pass MSB-first radix select over sortable keys. One workgroup per expert.
__global__ __launch_bounds__(1024)
void moe_select_threshold(const float* __restrict__ scoresT,
                          unsigned* __restrict__ Tkey, int* __restrict__ needOut,
                          int B, int M) {
  __shared__ unsigned hist[256];
  __shared__ unsigned bprefix, babove;

  const int e = blockIdx.x;
  const float* sc = scoresT + (size_t)e * B;

  unsigned prefix = 0, maskHi = 0, above = 0;

  for (int pass = 0; pass < 4; ++pass) {
    const int shift = 24 - pass * 8;
    for (int i = threadIdx.x; i < 256; i += blockDim.x) hist[i] = 0;
    __syncthreads();

    for (int t = threadIdx.x; t < B; t += blockDim.x) {
      unsigned k = sortKey(sc[t]);
      if ((k & maskHi) == prefix) atomicAdd(&hist[(k >> shift) & 255u], 1u);
    }
    __syncthreads();

    if (threadIdx.x == 0) {
      unsigned acc = above;
      unsigned b = 0;
      for (int i = 255; i >= 0; --i) {
        unsigned c = hist[i];
        if (acc + c >= (unsigned)M) { b = (unsigned)i; break; }
        acc += c;
      }
      bprefix = prefix | (b << shift);
      babove  = acc;
    }
    __syncthreads();
    prefix = bprefix;
    above  = babove;
    maskHi |= (0xFFu << shift);
  }

  if (threadIdx.x == 0) {
    Tkey[e]    = prefix;            // full 32-bit threshold key
    needOut[e] = M - (int)above;    // # of tie-keys to admit
  }
}

// ------------------- compact selected token indices ------------------------
__global__ __launch_bounds__(256)
void moe_compact(const float* __restrict__ scoresT, const unsigned* __restrict__ Tkey,
                 const int* __restrict__ need, int* __restrict__ idxArr,
                 int* __restrict__ mArr, int* __restrict__ cnt, int* __restrict__ cntEq,
                 int B, int M) {
  const int e = blockIdx.y;
  const int t = blockIdx.x * blockDim.x + threadIdx.x;
  if (t >= B) return;

  unsigned k = sortKey(scoresT[(size_t)e * B + t]);
  unsigned T = Tkey[e];
  if (k > T) {
    int pos = atomicAdd(&cnt[e], 1);
    idxArr[(size_t)e * M + pos] = t;
    atomicAdd(&mArr[t], 1);
  } else if (k == T) {
    int q = atomicAdd(&cntEq[e], 1);
    if (q < need[e]) {                 // cap ties so each expert takes exactly M
      int pos = atomicAdd(&cnt[e], 1);
      idxArr[(size_t)e * M + pos] = t;
      atomicAdd(&mArr[t], 1);
    }
  }
}

// ------------- fused expert MLP: gather -> GEMM1+ReLU -> GEMM2 -> scatter --
__global__ __launch_bounds__(128)
void moe_mlp(const float* __restrict__ feat, const _Float16* __restrict__ W1h,
             const float* __restrict__ b1, const _Float16* __restrict__ W2h,
             const float* __restrict__ b2, const int* __restrict__ idxArr,
             const int* __restrict__ mArr, float* __restrict__ out,
             int M, int tilesPerE) {
  __shared__ alignas(16) _Float16 Xs [TM][XS_STRIDE];  // gathered tokens (f16)
  __shared__ alignas(16) _Float16 W1s[KC][W1_STRIDE];  // W1 chunk, row-major over F
  __shared__ alignas(16) _Float16 W2s[O ][W2_STRIDE];  // W2 chunk, row-major over H-chunk
  __shared__ alignas(16) _Float16 Hs [TM][HC_STRIDE];  // hidden activations (f16)
  __shared__ float invmS[TM];
  __shared__ int   tokS [TM];

  const int bid  = blockIdx.x;
  const int e    = bid / tilesPerE;
  const int tb   = bid % tilesPerE;
  const int tid  = threadIdx.x;
  const int lane = tid & 31;
  const int wave = tid >> 5;     // 4 wave32's
  const int hh   = lane >> 4;    // lane half -> K sub-band
  const int lc   = lane & 15;    // M (A) / N (B,C,D) index within tile
  const int r0   = wave * 16;    // this wave's 16-row stripe of the 64-row tile

  // ---- gather X tile (two threads per token row; out-of-range rows get weight 0)
  {
    const int row   = tid >> 1;
    const int half  = tid & 1;
    const int slot  = tb * TM + row;
    const bool ok   = slot < M;
    const int src   = ok ? slot : 0;
    const int token = idxArr[(size_t)e * M + src];
    if (half == 0) {
      tokS[row] = token;
      int mm = mArr[token]; if (mm < 1) mm = 1;
      invmS[row] = ok ? (1.0f / (float)mm) : 0.0f;
    }
    const float* fr = feat + (size_t)token * F + half * (F / 2);
#pragma unroll
    for (int j = 0; j < F / 2; j += 4) {
      float4 x = *(const float4*)(fr + j);
      _Float16* d = &Xs[row][half * (F / 2) + j];
      d[0] = (_Float16)x.x; d[1] = (_Float16)x.y;
      d[2] = (_Float16)x.z; d[3] = (_Float16)x.w;
    }
  }

  // ---- Y accumulators (4 N-tiles over O=64), bias pre-loaded
  v8f acc[4];
#pragma unroll
  for (int nt = 0; nt < 4; ++nt) {
    float bv = b2[e * O + nt * 16 + lc];
#pragma unroll
    for (int v = 0; v < 8; ++v) acc[nt][v] = bv;
  }

  for (int c = 0; c < H / KC; ++c) {
    __syncthreads();   // previous iteration done with LDS (and gather visible)

    // ---- stage W1/W2 f16 chunks into LDS
    {
      const _Float16* gW1 = W1h + ((size_t)e * H + (size_t)c * KC) * F;  // [KC][F]
      const _Float16* gW2 = W2h + (size_t)e * O * H + c * KC;            // [O] rows, stride H
#if USE_ASYNC_LDS
#pragma unroll
      for (int it = 0; it < (KC * F) / (128 * 8); ++it) {   // 8 iters: 16KB
        int cid = tid + it * 128;                 // 16B chunk id, 16 chunks/row
        int row = cid >> 4;
        int col = (cid & 15) * 8;
        asyncCopy16(gW1 + (size_t)row * F + col, &W1s[row][col]);
      }
#pragma unroll
      for (int it = 0; it < (O * KC) / (128 * 8); ++it) {   // 4 iters: 8KB
        int cid = tid + it * 128;                 // 8 chunks/row
        int row = cid >> 3;
        int col = (cid & 7) * 8;
        asyncCopy16(gW2 + (size_t)row * H + col, &W2s[row][col]);
      }
      __builtin_amdgcn_s_wait_asynccnt(0);
#else
#pragma unroll
      for (int it = 0; it < (KC * F) / (128 * 8); ++it) {
        int cid = tid + it * 128;
        int row = cid >> 4;
        int col = (cid & 15) * 8;
        *(v8h*)&W1s[row][col] = *(const v8h*)(gW1 + (size_t)row * F + col);
      }
#pragma unroll
      for (int it = 0; it < (O * KC) / (128 * 8); ++it) {
        int cid = tid + it * 128;
        int row = cid >> 3;
        int col = (cid & 7) * 8;
        *(v8h*)&W2s[row][col] = *(const v8h*)(gW2 + (size_t)row * H + col);
      }
#endif
    }
    __syncthreads();

    // ---- GEMM1 + ReLU: Hs[r0..r0+15][0..KC) = relu(X @ W1c^T + b1c)
#pragma unroll
    for (int ht = 0; ht < KC / 16; ++ht) {
      float bv = b1[e * H + c * KC + ht * 16 + lc];
      v8f hacc;
#pragma unroll
      for (int v = 0; v < 8; ++v) hacc[v] = bv;
#pragma unroll
      for (int ks = 0; ks < F / 32; ++ks) {
        v16h a = ldfrag(&Xs[0][0],  r0 + lc,      XS_STRIDE, ks * 32, hh);
        v16h b = ldfrag(&W1s[0][0], ht * 16 + lc, W1_STRIDE, ks * 32, hh);
        hacc = __builtin_amdgcn_wmma_f32_16x16x32_f16(
            false, a, false, b, (short)0, hacc, false, false);
      }
#pragma unroll
      for (int v = 0; v < 8; ++v) {   // D layout: M = v + 8*hh, N = lc
        float x = hacc[v];
        x = x > 0.0f ? x : 0.0f;
        Hs[r0 + v + 8 * hh][ht * 16 + lc] = (_Float16)x;
      }
    }
    __syncthreads();

    // ---- GEMM2 accumulate: Y += Hc @ W2c^T
#pragma unroll
    for (int nt = 0; nt < 4; ++nt) {
#pragma unroll
      for (int ks = 0; ks < KC / 32; ++ks) {
        v16h a = ldfrag(&Hs[0][0],  r0 + lc,      HC_STRIDE, ks * 32, hh);
        v16h b = ldfrag(&W2s[0][0], nt * 16 + lc, W2_STRIDE, ks * 32, hh);
        acc[nt] = __builtin_amdgcn_wmma_f32_16x16x32_f16(
            false, a, false, b, (short)0, acc[nt], false, false);
      }
    }
  }

  // ---- scatter-add combine with 1/m weighting
#pragma unroll
  for (int nt = 0; nt < 4; ++nt) {
#pragma unroll
    for (int v = 0; v < 8; ++v) {
      const int row   = r0 + v + 8 * hh;
      const float w   = invmS[row];
      const int token = tokS[row];
      atomAddF(out + (size_t)token * O + nt * 16 + lc, acc[nt][v] * w);
    }
  }
}

// ---------------------------------------------------------------------------
extern "C" void kernel_launch(void* const* d_in, const int* in_sizes, int n_in,
                              void* d_out, int out_size, void* d_ws, size_t ws_size,
                              hipStream_t stream) {
  const float* feat = (const float*)d_in[0];
  const float* Wg   = (const float*)d_in[1];
  const float* bg   = (const float*)d_in[2];
  const float* W1   = (const float*)d_in[3];
  const float* b1   = (const float*)d_in[4];
  const float* W2   = (const float*)d_in[5];
  const float* b2   = (const float*)d_in[6];
  float* out = (float*)d_out;

  const int B = in_sizes[0] / F;                     // 262144
  long Ml = ((long)B + E - 1) / E;                   // ceil(CF*B/E), CF=1
  if (Ml > B) Ml = B;
  const int M = (int)Ml;                             // 16384

  // workspace layout (all offsets stay 16B aligned)
  char* ws = (char*)d_ws;
  size_t off = 0;
  float*    scoresT = (float*)   (ws + off); off += (size_t)E * B * sizeof(float);
  int*      idxArr  = (int*)     (ws + off); off += (size_t)E * M * sizeof(int);
  const size_t zoff = off;                            // zero-init region start
  int*      mArr    = (int*)     (ws + off); off += (size_t)B * sizeof(int);
  int*      cnt     = (int*)     (ws + off); off += (size_t)E * sizeof(int);
  int*      cntEq   = (int*)     (ws + off); off += (size_t)E * sizeof(int);
  const size_t zbytes = off - zoff;
  unsigned* Tkey    = (unsigned*)(ws + off); off += (size_t)E * sizeof(unsigned);
  int*      need    = (int*)     (ws + off); off += (size_t)E * sizeof(int);
  _Float16* W1h     = (_Float16*)(ws + off); off += (size_t)E * H * F * sizeof(_Float16);
  _Float16* W2h     = (_Float16*)(ws + off); off += (size_t)E * O * H * sizeof(_Float16);
  (void)ws_size; (void)n_in;

  (void)hipMemsetAsync(out, 0, (size_t)out_size * sizeof(float), stream);
  (void)hipMemsetAsync(ws + zoff, 0, zbytes, stream);

  const int n1q = E * H * F / 4;
  const int n2q = E * O * H / 4;
  moe_cvt_weights<<<dim3((n1q + n2q + 255) / 256), dim3(256), 0, stream>>>(
      W1, W2, W1h, W2h, n1q, n2q);

  moe_gate<<<dim3((B + 255) / 256), dim3(256), 0, stream>>>(feat, Wg, bg, scoresT, B);
  moe_select_threshold<<<dim3(E), dim3(1024), 0, stream>>>(scoresT, Tkey, need, B, M);
  moe_compact<<<dim3((B + 255) / 256, E), dim3(256), 0, stream>>>(
      scoresT, Tkey, need, idxArr, mArr, cnt, cntEq, B, M);

  const int tilesPerE = (M + TM - 1) / TM;
  moe_mlp<<<dim3(E * tilesPerE), dim3(128), 0, stream>>>(
      feat, W1h, b1, W2h, b2, idxArr, mArr, out, M, tilesPerE);
}
